// NeuralSynth_37529424233042
// MI455X (gfx1250) — compile-verified, matrix operand order checked by reading, not run on hardware
//
#include <hip/hip_runtime.h>
#include <hip/hip_bf16.h>

#define HIDN 512
#define NPART 100
#define FSZ 160
#define SEQL 200
#define BSZ 16
#define TLEN 32000
#define NBIN 81
#define ROWS 3200   /* BSZ*SEQL */

typedef __attribute__((ext_vector_type(16))) _Float16 v16h;
typedef __attribute__((ext_vector_type(8)))  float    v8f;
typedef __attribute__((ext_vector_type(2)))  _Float16 h2;

// ---------------------------------------------------------------- conversions
__global__ void cvt_f32_f16_kernel(const float* __restrict__ src, _Float16* __restrict__ dst, int n) {
    int i = blockIdx.x * blockDim.x + threadIdx.x;
    if (i < n) dst[i] = (_Float16)src[i];
}

// W is (K x N) row-major; Wt becomes (N x K) row-major so B fragments load contiguous k-pairs
__global__ void transpose_cvt_kernel(const float* __restrict__ W, _Float16* __restrict__ Wt, int K, int N) {
    int i = blockIdx.x * blockDim.x + threadIdx.x;
    if (i >= K * N) return;
    int k = i / N, n = i % N;
    Wt[(size_t)n * K + k] = (_Float16)W[i];
}

// ------------------------------------------------------------ WMMA GEMM + bias
// C(MxN) = A(MxK,f16,row-major) * B^T  with B stored as (N x K) f16 row-major.
__global__ void wmma_gemm_kernel(const _Float16* __restrict__ A, const _Float16* __restrict__ B,
                                 const float* __restrict__ bias, float* __restrict__ C,
                                 int M, int N, int K) {
    int gw   = (blockIdx.x * blockDim.x + threadIdx.x) >> 5;
    int lane = threadIdx.x & 31;
    int tiles_n = N >> 4, tiles_m = M >> 4;
    if (gw >= tiles_m * tiles_n) return;
    int tm = gw / tiles_n, tn = gw % tiles_n;
    int arow = tm * 16 + (lane & 15);
    int bcol = tn * 16 + (lane & 15);
    const int koff = (lane & 16) ? 8 : 0;
    v8f acc; for (int i = 0; i < 8; ++i) acc[i] = 0.f;
    for (int k0 = 0; k0 < K; k0 += 32) {
        __builtin_prefetch(A + (size_t)arow * K + k0 + 64, 0, 1);
        v16h a, b;
#pragma unroll
        for (int v = 0; v < 8; ++v) {
            int kk = k0 + koff + ((v < 4) ? 2 * v : 16 + 2 * (v - 4));
            h2 pa = *(const h2*)(A + (size_t)arow * K + kk);
            h2 pb = *(const h2*)(B + (size_t)bcol * K + kk);
            a[2 * v] = pa.x; a[2 * v + 1] = pa.y;
            b[2 * v] = pb.x; b[2 * v + 1] = pb.y;
        }
        acc = __builtin_amdgcn_wmma_f32_16x16x32_f16(false, a, false, b, (short)0, acc, false, false);
    }
    int n  = tn * 16 + (lane & 15);
    float bv = bias ? bias[n] : 0.f;
    int mb = tm * 16 + ((lane & 16) ? 8 : 0);
#pragma unroll
    for (int v = 0; v < 8; ++v) {
        int m = mb + v;
        if (m < M) C[(size_t)m * N + n] = acc[v] + bv;
    }
}

// -------------------------------------------------- LayerNorm + ReLU epilogue
__global__ void ln_relu_kernel(const float* __restrict__ X, const float* __restrict__ g,
                               const float* __restrict__ be, float* __restrict__ Yf,
                               _Float16* __restrict__ Y16, int ld16, int off16) {
    int row = blockIdx.x, tid = threadIdx.x;
    const float* x = X + (size_t)row * HIDN;
    float a0 = x[tid], a1 = x[tid + 256];
    __shared__ float s1[256], s2[256];
    s1[tid] = a0 + a1; s2[tid] = a0 * a0 + a1 * a1;
    __syncthreads();
    for (int st = 128; st > 0; st >>= 1) {
        if (tid < st) { s1[tid] += s1[tid + st]; s2[tid] += s2[tid + st]; }
        __syncthreads();
    }
    float mean = s1[0] * (1.f / HIDN);
    float var  = s2[0] * (1.f / HIDN) - mean * mean;
    float rs   = rsqrtf(var + 1e-5f);
    float y0 = fmaxf((a0 - mean) * rs * g[tid]       + be[tid],       0.f);
    float y1 = fmaxf((a1 - mean) * rs * g[tid + 256] + be[tid + 256], 0.f);
    Yf[(size_t)row * HIDN + tid]       = y0;
    Yf[(size_t)row * HIDN + tid + 256] = y1;
    _Float16* y16 = Y16 + (size_t)row * ld16 + off16;
    y16[tid] = (_Float16)y0; y16[tid + 256] = (_Float16)y1;
}

// ------------------------------------- first MLP layer (1 -> 512) + LN + ReLU
__global__ void first_layer_kernel(const float* __restrict__ xin, const float* __restrict__ W0,
                                   const float* __restrict__ b0, const float* __restrict__ g,
                                   const float* __restrict__ be, float* __restrict__ Yf,
                                   _Float16* __restrict__ Y16) {
    int row = blockIdx.x, tid = threadIdx.x;
    float xv = xin[row];
    float a0 = xv * W0[tid] + b0[tid];
    float a1 = xv * W0[tid + 256] + b0[tid + 256];
    __shared__ float s1[256], s2[256];
    s1[tid] = a0 + a1; s2[tid] = a0 * a0 + a1 * a1;
    __syncthreads();
    for (int st = 128; st > 0; st >>= 1) {
        if (tid < st) { s1[tid] += s1[tid + st]; s2[tid] += s2[tid + st]; }
        __syncthreads();
    }
    float mean = s1[0] * (1.f / HIDN);
    float var  = s2[0] * (1.f / HIDN) - mean * mean;
    float rs   = rsqrtf(var + 1e-5f);
    float y0 = fmaxf((a0 - mean) * rs * g[tid]       + be[tid],       0.f);
    float y1 = fmaxf((a1 - mean) * rs * g[tid + 256] + be[tid + 256], 0.f);
    Yf[(size_t)row * HIDN + tid]       = y0;
    Yf[(size_t)row * HIDN + tid + 256] = y1;
    _Float16* y16 = Y16 + (size_t)row * HIDN;
    y16[tid] = (_Float16)y0; y16[tid + 256] = (_Float16)y1;
}

// --------------------------------------- persistent GRU scan (1 WG, 16 waves)
// Each wave owns 32 hidden units -> 6 WMMA output tiles (r,z,n x 2). M=16=batch.
__global__ void __launch_bounds__(512) gru_scan_kernel(const float* __restrict__ gx,
        const _Float16* __restrict__ Whh, const float* __restrict__ bhh,
        float* __restrict__ hseq, _Float16* __restrict__ hseq16) {
    __shared__ _Float16 hsh[16 * HIDN];
    int tid = threadIdx.x, lane = tid & 31, wave = tid >> 5;
    for (int i = tid; i < 16 * HIDN; i += 512) hsh[i] = (_Float16)0.f;
    __syncthreads();
    const int jbase0 = wave * 32;
    const int koff = (lane & 16) ? 8 : 0;
    const int l15  = lane & 15;
    const int mofs = (lane & 16) ? 8 : 0;
    for (int s = 0; s < SEQL; ++s) {
        v8f acc[3][2];
        for (int g2 = 0; g2 < 3; ++g2)
            for (int t2 = 0; t2 < 2; ++t2)
                for (int i = 0; i < 8; ++i) acc[g2][t2][i] = 0.f;
        for (int k0 = 0; k0 < HIDN; k0 += 32) {
            v16h a;
#pragma unroll
            for (int v = 0; v < 8; ++v) {
                int kk = k0 + koff + ((v < 4) ? 2 * v : 16 + 2 * (v - 4));
                h2 pa = *(const h2*)&hsh[l15 * HIDN + kk];
                a[2 * v] = pa.x; a[2 * v + 1] = pa.y;
            }
#pragma unroll
            for (int g2 = 0; g2 < 3; ++g2)
#pragma unroll
                for (int t2 = 0; t2 < 2; ++t2) {
                    int ncol = g2 * HIDN + jbase0 + t2 * 16 + l15;
                    v16h b;
#pragma unroll
                    for (int v = 0; v < 8; ++v) {
                        int kk = k0 + koff + ((v < 4) ? 2 * v : 16 + 2 * (v - 4));
                        h2 pb = *(const h2*)(Whh + (size_t)ncol * HIDN + kk);
                        b[2 * v] = pb.x; b[2 * v + 1] = pb.y;
                    }
                    acc[g2][t2] = __builtin_amdgcn_wmma_f32_16x16x32_f16(
                        false, a, false, b, (short)0, acc[g2][t2], false, false);
                }
        }
        float hnew[2][8];
#pragma unroll
        for (int t2 = 0; t2 < 2; ++t2) {
            int j = jbase0 + t2 * 16 + l15;
#pragma unroll
            for (int v = 0; v < 8; ++v) {
                int m = v + mofs;
                const float* gxr = gx + ((size_t)m * SEQL + s) * (3 * HIDN);
                float ghr = acc[0][t2][v] + bhh[j];
                float ghz = acc[1][t2][v] + bhh[HIDN + j];
                float ghn = acc[2][t2][v] + bhh[2 * HIDN + j];
                float r  = 1.f / (1.f + __expf(-(gxr[j] + ghr)));
                float z  = 1.f / (1.f + __expf(-(gxr[HIDN + j] + ghz)));
                float nn = tanhf(gxr[2 * HIDN + j] + r * ghn);
                float hold = (float)hsh[m * HIDN + j];
                hnew[t2][v] = (1.f - z) * nn + z * hold;
            }
        }
        __syncthreads();
#pragma unroll
        for (int t2 = 0; t2 < 2; ++t2) {
            int j = jbase0 + t2 * 16 + l15;
#pragma unroll
            for (int v = 0; v < 8; ++v) {
                int m = v + mofs;
                float hv = hnew[t2][v];
                hsh[m * HIDN + j] = (_Float16)hv;
                size_t o = ((size_t)m * SEQL + s) * HIDN + j;
                hseq[o]   = hv;
                hseq16[o] = (_Float16)hv;
            }
        }
        __syncthreads();
    }
}

// ------------------------------------------------------- output heads (VALU)
__global__ void heads_kernel(const float* __restrict__ X,
        const float* Wa, const float* ba, const float* Wal, const float* bal,
        const float* Wf, const float* bf,
        float* __restrict__ amp, float* __restrict__ alpha, float* __restrict__ filt) {
    int idx = blockIdx.x * blockDim.x + threadIdx.x;
    if (idx >= ROWS * 182) return;
    int row = idx / 182, c = idx % 182;
    const float* x = X + (size_t)row * HIDN;
    const float* w; int stride; float acc;
    if (c == 0)        { w = Wa;            stride = 1;     acc = ba[0]; }
    else if (c < 101)  { w = Wal + (c - 1); stride = NPART; acc = bal[c - 1]; }
    else               { w = Wf + (c - 101); stride = NBIN; acc = bf[c - 101]; }
    for (int k = 0; k < HIDN; ++k) acc += x[k] * w[(size_t)k * stride];
    float sg = 1.f / (1.f + __expf(-acc));
    float ms = 2.f * __powf(sg, 2.302585092994046f) + 1e-7f;
    if (c == 0)       amp[row] = ms;
    else if (c < 101) alpha[(size_t)row * NPART + (c - 1)] = ms;
    else              filt[(size_t)row * NBIN + (c - 101)] = ms;
}

__global__ void alpha_norm_kernel(float* __restrict__ alpha) {
    int row = blockIdx.x * blockDim.x + threadIdx.x;
    if (row >= ROWS) return;
    float* a = alpha + (size_t)row * NPART;
    float s = 0.f;
    for (int p = 0; p < NPART; ++p) s += a[p];
    float inv = 1.f / s;
    for (int p = 0; p < NPART; ++p) a[p] *= inv;
}

// ------------------------------------------- phase accumulation (block scan)
__global__ void phase_scan_kernel(const float* __restrict__ f0, float* __restrict__ phi) {
    int b = blockIdx.x;
    int tid = threadIdx.x, lane = tid & 31, wid = tid >> 5;
    __shared__ float wsum[8];
    __shared__ float carry_s;
    if (tid == 0) carry_s = 0.f;
    __syncthreads();
    float f0first = f0[b * SEQL] * (1.f / 16000.f);
    for (int base = 0; base < TLEN; base += 256) {
        int t = base + tid;
        float coord = ((float)t + 0.5f) * (1.f / 160.f) - 0.5f;
        coord = fminf(fmaxf(coord, 0.f), (float)(SEQL - 1));
        int lo = (int)coord;
        int hi = min(lo + 1, SEQL - 1);
        float w = coord - (float)lo;
        float f = (f0[b * SEQL + lo] * (1.f - w) + f0[b * SEQL + hi] * w) * (1.f / 16000.f);
        float v = f;
        for (int off = 1; off < 32; off <<= 1) {
            float u = __shfl_up(v, off, 32);
            if (lane >= off) v += u;
        }
        if (lane == 31) wsum[wid] = v;
        __syncthreads();
        if (tid == 0) {
            float s2 = 0.f;
            for (int i = 0; i < 8; ++i) { float tmp = wsum[i]; wsum[i] = s2; s2 += tmp; }
        }
        __syncthreads();
        float carry = carry_s;
        float incl = v + wsum[wid] + carry;
        phi[(size_t)b * TLEN + t] = 6.283185307179586f * (incl - f0first);
        __syncthreads();
        if (tid == 255) carry_s = incl;
        __syncthreads();
    }
}

// ---------------------------------- harmonic synthesis + upsampled cond writes
__global__ void harmonic_kernel(const float* __restrict__ phi, const float* __restrict__ amp,
                                const float* __restrict__ alpha, float* __restrict__ ampu,
                                float* __restrict__ alphau, float* __restrict__ ybuf) {
    int idx = blockIdx.x * blockDim.x + threadIdx.x;
    if (idx >= BSZ * TLEN) return;
    int b = idx / TLEN, t = idx % TLEN;
    float coord = ((float)t + 0.5f) * (1.f / 160.f) - 0.5f;
    coord = fminf(fmaxf(coord, 0.f), (float)(SEQL - 1));
    int lo = (int)coord;
    int hi = min(lo + 1, SEQL - 1);
    float w = coord - (float)lo;
    float av = amp[b * SEQL + lo] * (1.f - w) + amp[b * SEQL + hi] * w;
    const float inv2pi = 0.15915494309189535f, twopi = 6.283185307179586f;
    float ph = phi[idx];
    ph -= twopi * rintf(ph * inv2pi);
    const float* alo = alpha + ((size_t)b * SEQL + lo) * NPART;
    const float* ahi = alpha + ((size_t)b * SEQL + hi) * NPART;
    float* ao = alphau + (size_t)idx * NPART;
    float accv = 0.f;
    for (int p = 0; p < NPART; ++p) {
        float al = alo[p] * (1.f - w) + ahi[p] * w;
        ao[p] = al;
        float arg = ph * (float)(p + 1);
        arg -= twopi * rintf(arg * inv2pi);
        accv += al * __sinf(arg);
    }
    ampu[idx] = av;
    ybuf[idx] = av * accv;
}

// ------------------------------- noise filter: A = rfft(win * irfft(.)) matrix
__global__ void amat_kernel(float2* __restrict__ A) {
    int idx = blockIdx.x * blockDim.x + threadIdx.x;
    if (idx >= NBIN * NBIN) return;
    int kp = idx / NBIN, k = idx % NBIN;
    float wk = (k == 0 || k == NBIN - 1) ? 1.f : 2.f;
    float ar = 0.f, ai = 0.f;
    const float c0 = 6.283185307179586f / 160.f;
    for (int n = 0; n < FSZ; ++n) {
        float win = 0.5f * (1.f + __cosf(c0 * n));          // roll(hann, 80)
        float ck  = (wk * (1.f / 160.f)) * __cosf(c0 * k * n);
        float sy, cy; __sincosf(c0 * kp * n, &sy, &cy);
        ar += win * ck * cy;
        ai -= win * ck * sy;
    }
    A[idx] = make_float2(ar, ai);
}

__global__ void noise_dft_kernel(const float* __restrict__ noise, float2* __restrict__ Sn) {
    int idx = blockIdx.x * blockDim.x + threadIdx.x;
    if (idx >= ROWS * NBIN) return;
    int f = idx / NBIN, k = idx % NBIN;
    const float* x = noise + (size_t)f * FSZ;
    float step = -6.283185307179586f * (float)k / 160.f;
    float re = 0.f, im = 0.f;
    for (int n = 0; n < FSZ; ++n) {
        float s, c; __sincosf(step * n, &s, &c);
        re += x[n] * c; im += x[n] * s;
    }
    Sn[idx] = make_float2(re, im);
}

__global__ void sfilt_kernel(const float* __restrict__ fc, const float2* __restrict__ A,
                             const float2* __restrict__ Sn, float2* __restrict__ Sout) {
    int idx = blockIdx.x * blockDim.x + threadIdx.x;
    if (idx >= ROWS * NBIN) return;
    int f = idx / NBIN, kp = idx % NBIN;
    const float2* Ar = A + (size_t)kp * NBIN;
    const float* fcr = fc + (size_t)f * NBIN;
    float hr = 0.f, hi = 0.f;
    for (int k = 0; k < NBIN; ++k) { hr += Ar[k].x * fcr[k]; hi += Ar[k].y * fcr[k]; }
    float2 sn = Sn[idx];
    Sout[idx] = make_float2(hr * sn.x - hi * sn.y, hr * sn.y + hi * sn.x);
}

__global__ void noise_irfft_add_kernel(const float2* __restrict__ Sf, float* __restrict__ y) {
    int idx = blockIdx.x * blockDim.x + threadIdx.x;
    if (idx >= ROWS * FSZ) return;
    int f = idx / FSZ, n = idx % FSZ;
    const float2* S = Sf + (size_t)f * NBIN;
    const float c0 = 6.283185307179586f / 160.f;
    float acc = 0.f;
    for (int k = 0; k < NBIN; ++k) {
        float wk = (k == 0 || k == NBIN - 1) ? 1.f : 2.f;
        float s, c; __sincosf(c0 * k * n, &s, &c);
        acc += wk * (S[k].x * c - S[k].y * s);
    }
    y[idx] += acc * (1.f / 160.f);
}

// ----------------------------------------------- reverb: mixed-radix Stockham
__global__ void fft_pack_kernel(const float* __restrict__ y, const float* __restrict__ imp,
                                float2* __restrict__ c) {
    int idx = blockIdx.x * blockDim.x + threadIdx.x;
    if (idx >= 17 * TLEN) return;
    int b = idx / TLEN, t = idx % TLEN;
    float v = (b < 16) ? y[idx] : tanhf(imp[t]);
    c[idx] = make_float2(v, 0.f);
}

__global__ void fft_stage_kernel(const float2* __restrict__ x, float2* __restrict__ y,
                                 int n, int s, int radix, int batch) {
    int per = TLEN / radix;
    int idx = blockIdx.x * blockDim.x + threadIdx.x;
    if (idx >= batch * per) return;
    int b = idx / per, r2 = idx % per;
    int q = r2 % s, p = r2 / s;
    int m = n / radix;
    const float2* xb = x + (size_t)b * TLEN;
    float2* yb = y + (size_t)b * TLEN;
    if (radix == 2) {
        float2 a0 = xb[q + s * p], a1 = xb[q + s * (p + m)];
        float sn, cs; __sincosf(-6.283185307179586f * (float)p / (float)n, &sn, &cs);
        yb[q + s * (2 * p)] = make_float2(a0.x + a1.x, a0.y + a1.y);
        float dr = a0.x - a1.x, di = a0.y - a1.y;
        yb[q + s * (2 * p + 1)] = make_float2(dr * cs - di * sn, dr * sn + di * cs);
    } else { // radix 5
        float2 a[5];
#pragma unroll
        for (int r = 0; r < 5; ++r) a[r] = xb[q + s * (p + m * r)];
        const float w5r[5] = {1.f, 0.30901699437494745f, -0.8090169943749475f,
                              -0.8090169943749475f, 0.30901699437494745f};
        const float w5i[5] = {0.f, -0.9510565162951535f, -0.5877852522924731f,
                              0.5877852522924731f, 0.9510565162951535f};
#pragma unroll
        for (int u = 0; u < 5; ++u) {
            float br = 0.f, bi = 0.f;
#pragma unroll
            for (int r = 0; r < 5; ++r) {
                int e = (u * r) % 5;
                br += a[r].x * w5r[e] - a[r].y * w5i[e];
                bi += a[r].x * w5i[e] + a[r].y * w5r[e];
            }
            float sn, cs;
            __sincosf(-6.283185307179586f * (float)(p * u) / (float)n, &sn, &cs);
            yb[q + s * (5 * p + u)] = make_float2(br * cs - bi * sn, br * sn + bi * cs);
        }
    }
}

__global__ void specmul_conj_kernel(const float2* __restrict__ S, float2* __restrict__ Z) {
    int idx = blockIdx.x * blockDim.x + threadIdx.x;
    if (idx >= 16 * TLEN) return;
    int t = idx % TLEN;
    float2 a = S[idx];
    float2 ir = S[(size_t)16 * TLEN + t];
    float re = a.x * ir.x - a.y * ir.y;
    float im = a.x * ir.y + a.y * ir.x;
    Z[idx] = make_float2(re, -im);     // conj for inverse-via-forward-FFT trick
}

__global__ void extract_real_kernel(const float2* __restrict__ Z, float* __restrict__ out) {
    int idx = blockIdx.x * blockDim.x + threadIdx.x;
    if (idx >= 16 * TLEN) return;
    out[idx] = Z[idx].x * (1.f / 32000.f);
}

// ============================================================================
extern "C" void kernel_launch(void* const* d_in, const int* in_sizes, int n_in,
                              void* d_out, int out_size, void* d_ws, size_t ws_size,
                              hipStream_t stream) {
    if (n_in < 50) return;
    auto F = [&](int i) { return (const float*)d_in[i]; };

    const float* f0   = F(0);
    const float* lo   = F(1);
    // params flattened: f0_mlp(2..13) lo_mlp(14..25) gru(26..29) fi_mlp(30..41)
    // amp(42,43) alpha(44,45) filter(46,47) noise(48) impulse(49)
    const float* noise   = F(48);
    const float* impulse = F(49);

    // ---- workspace bump allocator ----
    char* wsp = (char*)d_ws;
    size_t off = 0;
    auto alloc = [&](size_t bytes) -> void* {
        void* p = wsp + off;
        off = (off + bytes + 255) & ~(size_t)255;
        return p;
    };
    float*     actA   = (float*)alloc((size_t)ROWS * HIDN * 4);
    float*     actB   = (float*)alloc((size_t)ROWS * HIDN * 4);
    _Float16*  a16A   = (_Float16*)alloc((size_t)ROWS * HIDN * 2);
    _Float16*  xcat16 = (_Float16*)alloc((size_t)ROWS * 2 * HIDN * 2);
    _Float16*  wslot  = (_Float16*)alloc((size_t)3 * HIDN * 2 * HIDN * 2); // 1536x1024 f16
    _Float16*  whh16  = (_Float16*)alloc((size_t)3 * HIDN * HIDN * 2);
    float*     gx     = (float*)alloc((size_t)ROWS * 3 * HIDN * 4);
    float*     hseq   = (float*)alloc((size_t)ROWS * HIDN * 4);
    _Float16*  hseq16 = (_Float16*)alloc((size_t)ROWS * HIDN * 2);
    float*     ampb   = (float*)alloc((size_t)ROWS * 4);
    float*     alphab = (float*)alloc((size_t)ROWS * NPART * 4);
    float*     filtb  = (float*)alloc((size_t)ROWS * NBIN * 4);
    float*     phi    = (float*)alloc((size_t)BSZ * TLEN * 4);
    float*     ybuf   = (float*)alloc((size_t)BSZ * TLEN * 4);
    float2*    Amat   = (float2*)alloc((size_t)NBIN * NBIN * 8);
    float2*    Snoise = (float2*)alloc((size_t)ROWS * NBIN * 8);
    float2*    cA     = (float2*)alloc((size_t)17 * TLEN * 8);
    float2*    cB     = (float2*)alloc((size_t)17 * TLEN * 8);
    (void)ws_size;

    // ---- output layout ----
    float* out        = (float*)d_out;
    float* out_yconv  = out;
    float* out_ampu   = out + (size_t)BSZ * TLEN;
    float* out_alphau = out + (size_t)2 * BSZ * TLEN;
    float* out_S      = out + (size_t)2 * BSZ * TLEN + (size_t)BSZ * TLEN * NPART;
    float2* Sfilt     = (float2*)out_S;
    (void)out_size; (void)in_sizes;

    #define NB256(n) (((n) + 255) / 256)

    // ---- GEMM helper (one 512->512 MLP layer: transpose weight, GEMM, LN) ----
    auto mlp_layer = [&](const _Float16* a16, int wi, int bi, int gi, int bei,
                         _Float16* dst16, int ld16, int off16) {
        transpose_cvt_kernel<<<NB256(HIDN * HIDN), 256, 0, stream>>>(F(wi), wslot, HIDN, HIDN);
        int waves = (ROWS / 16) * (HIDN / 16);
        wmma_gemm_kernel<<<(waves + 7) / 8, 256, 0, stream>>>(a16, wslot, F(bi), actB,
                                                              ROWS, HIDN, HIDN);
        ln_relu_kernel<<<ROWS, 256, 0, stream>>>(actB, F(gi), F(bei), actA, dst16, ld16, off16);
    };

    // ---- f0 branch ----
    first_layer_kernel<<<ROWS, 256, 0, stream>>>(f0, F(2), F(3), F(4), F(5), actA, a16A);
    mlp_layer(a16A, 6, 7, 8, 9, a16A, HIDN, 0);
    mlp_layer(a16A, 10, 11, 12, 13, xcat16, 2 * HIDN, 0);
    // ---- lo branch ----
    first_layer_kernel<<<ROWS, 256, 0, stream>>>(lo, F(14), F(15), F(16), F(17), actA, a16A);
    mlp_layer(a16A, 18, 19, 20, 21, a16A, HIDN, 0);
    mlp_layer(a16A, 22, 23, 24, 25, xcat16, 2 * HIDN, HIDN);

    // ---- GRU input projection: gx = xcat @ Wih^T + bih ----
    cvt_f32_f16_kernel<<<NB256(3 * HIDN * 2 * HIDN), 256, 0, stream>>>(F(26), wslot,
                                                                       3 * HIDN * 2 * HIDN);
    {
        int waves = (ROWS / 16) * ((3 * HIDN) / 16);
        wmma_gemm_kernel<<<(waves + 7) / 8, 256, 0, stream>>>(xcat16, wslot, F(28), gx,
                                                              ROWS, 3 * HIDN, 2 * HIDN);
    }
    // ---- GRU recurrence (persistent WMMA scan) ----
    cvt_f32_f16_kernel<<<NB256(3 * HIDN * HIDN), 256, 0, stream>>>(F(27), whh16, 3 * HIDN * HIDN);
    gru_scan_kernel<<<1, 512, 0, stream>>>(gx, whh16, F(29), hseq, hseq16);

    // ---- fi_mlp ----
    mlp_layer(hseq16, 30, 31, 32, 33, a16A, HIDN, 0);
    mlp_layer(a16A, 34, 35, 36, 37, a16A, HIDN, 0);
    mlp_layer(a16A, 38, 39, 40, 41, a16A, HIDN, 0);

    // ---- heads + alpha normalization ----
    heads_kernel<<<NB256(ROWS * 182), 256, 0, stream>>>(actA, F(42), F(43), F(44), F(45),
                                                        F(46), F(47), ampb, alphab, filtb);
    alpha_norm_kernel<<<NB256(ROWS), 256, 0, stream>>>(alphab);

    // ---- phase accumulation + harmonic synthesis ----
    phase_scan_kernel<<<BSZ, 256, 0, stream>>>(f0, phi);
    harmonic_kernel<<<NB256(BSZ * TLEN), 256, 0, stream>>>(phi, ampb, alphab,
                                                           out_ampu, out_alphau, ybuf);

    // ---- noise filtering ----
    amat_kernel<<<NB256(NBIN * NBIN), 256, 0, stream>>>(Amat);
    noise_dft_kernel<<<NB256(ROWS * NBIN), 256, 0, stream>>>(noise, Snoise);
    sfilt_kernel<<<NB256(ROWS * NBIN), 256, 0, stream>>>(filtb, Amat, Snoise, Sfilt);
    noise_irfft_add_kernel<<<NB256(ROWS * FSZ), 256, 0, stream>>>(Sfilt, ybuf);

    // ---- reverb: forward FFT (17 signals), spectral mult, inverse FFT ----
    fft_pack_kernel<<<NB256(17 * TLEN), 256, 0, stream>>>(ybuf, impulse, cA);
    const int RAD[11] = {5, 5, 5, 2, 2, 2, 2, 2, 2, 2, 2};
    float2 *cur = cA, *nxt = cB;
    int nn = TLEN, ss = 1;
    for (int st = 0; st < 11; ++st) {
        int R = RAD[st];
        int tot = 17 * (TLEN / R);
        fft_stage_kernel<<<NB256(tot), 256, 0, stream>>>(cur, nxt, nn, ss, R, 17);
        nn /= R; ss *= R;
        float2* tp = cur; cur = nxt; nxt = tp;
    }
    specmul_conj_kernel<<<NB256(16 * TLEN), 256, 0, stream>>>(cur, nxt);
    { float2* tp = cur; cur = nxt; nxt = tp; }
    nn = TLEN; ss = 1;
    for (int st = 0; st < 11; ++st) {
        int R = RAD[st];
        int tot = 16 * (TLEN / R);
        fft_stage_kernel<<<NB256(tot), 256, 0, stream>>>(cur, nxt, nn, ss, R, 16);
        nn /= R; ss *= R;
        float2* tp = cur; cur = nxt; nxt = tp;
    }
    extract_real_kernel<<<NB256(16 * TLEN), 256, 0, stream>>>(cur, out_yconv);
}
